// TemporalGNNWithMemory_43121471652513
// MI455X (gfx1250) — matrix operand order repaired and checked by reading, not compiled, not run on hardware
//
#include <hip/hip_runtime.h>

// ---------------------------------------------------------------------------
// Problem constants (match reference)
// ---------------------------------------------------------------------------
#define NN    50000      // nodes
#define EE    800000     // edges
#define DIN   256
#define HID   128
#define MEMN  64
#define CLS   4
#define BAT   128
#define NHEAD 8
#define HDIM  16

typedef __bf16 bf16;
typedef __attribute__((ext_vector_type(16))) __bf16 v16bf;
typedef __attribute__((ext_vector_type(8)))  __bf16 v8bf;
typedef __attribute__((ext_vector_type(8)))  float  v8f;

union AFrag { v16bf v; v8bf h[2]; };

__device__ __forceinline__ void atomAddF(float* p, float v) {
    (void)__hip_atomic_fetch_add(p, v, __ATOMIC_RELAXED, __HIP_MEMORY_SCOPE_AGENT);
}

// ---------------------------------------------------------------------------
// Casting helpers
// ---------------------------------------------------------------------------
__global__ void k_cast_bf16(const float* __restrict__ src, bf16* __restrict__ dst, int n) {
    int i = blockIdx.x * 256 + threadIdx.x;
    if (i < n) dst[i] = (bf16)src[i];
}

// W is [K,N] row-major (as used in x@W). Produce Wt [N,K] bf16 so that the
// WMMA B-fragment (per-lane column, contiguous K) is a single 32B load.
__global__ void k_transpose_cast(const float* __restrict__ W, bf16* __restrict__ Wt,
                                 int K, int N) {
    int i = blockIdx.x * 256 + threadIdx.x;
    if (i < K * N) {
        int k = i / N, n = i % N;
        Wt[(long)n * K + k] = (bf16)W[i];
    }
}

// ---------------------------------------------------------------------------
// Strip WMMA GEMM, specialized for Ncols == 128 (all GEMMs here).
// One wave computes a full 16x128 output strip: the A fragment is loaded once
// per K-step and reused by 8 independent V_WMMA_F32_16X16X32_BF16 ops with 8
// separate f32 accumulators (no D->A/B hazards; deep XDL pipelining).
//
//   mode 0: C = relu(A@B + bias)                     (feature transform)
//   mode 1: C = A@B (raw xw), D = gb + C*dis[row]^2  (GCN bias + self-loop)
// ---------------------------------------------------------------------------
__global__ void k_wmma_gemm_strip(const bf16* __restrict__ A, const bf16* __restrict__ Bt,
                                  float* __restrict__ C, float* __restrict__ D,
                                  int Mrows, int K,
                                  const float* __restrict__ bias,
                                  const float* __restrict__ dis, int mode) {
    int wave  = blockIdx.x * (blockDim.x >> 5) + (threadIdx.x >> 5);
    int mt    = Mrows >> 4;
    if (wave >= mt) return;                    // wave-uniform: EXEC stays all-1s
    int lane  = threadIdx.x & 31;
    int half  = lane >> 4, lo = lane & 15;

    const bf16* arow = A  + (long)(wave * 16 + lo) * K;  // A: lane row = M
    const bf16* bcol = Bt + (long)lo * K + half * 16;    // B: lane row = N column

    v8f c[8];
#pragma unroll
    for (int t = 0; t < 8; ++t) c[t] = (v8f){};

    for (int kk = 0; kk < K; kk += 32) {
        AFrag a;
        a.h[0] = *(const v8bf*)(arow + kk + half * 16);      // K = kk+half*16 .. +7
        a.h[1] = *(const v8bf*)(arow + kk + half * 16 + 8);  // K = kk+half*16+8 ..
#pragma unroll
        for (int t = 0; t < 8; ++t) {
            v16bf b = *(const v16bf*)(bcol + (long)t * 16 * K + kk);
            c[t] = __builtin_amdgcn_wmma_f32_16x16x32_bf16(
                       false, a.v, false, b, (short)0, c[t], false, false);
        }
    }

    if (mode == 0) {
#pragma unroll
        for (int t = 0; t < 8; ++t) {
            int colg = t * 16 + lo;
            float bb = bias[colg];
#pragma unroll
            for (int r = 0; r < 8; ++r) {
                int rowg = wave * 16 + half * 8 + r;    // C/D layout: row = half*8+r
                C[(long)rowg * HID + colg] = fmaxf(c[t][r] + bb, 0.0f);
            }
        }
    } else {
        float sn2[8];
#pragma unroll
        for (int r = 0; r < 8; ++r) {
            float s = dis[wave * 16 + half * 8 + r];
            sn2[r] = s * s;
        }
#pragma unroll
        for (int t = 0; t < 8; ++t) {
            int colg = t * 16 + lo;
            float bb = bias[colg];                       // gb
#pragma unroll
            for (int r = 0; r < 8; ++r) {
                int rowg = wave * 16 + half * 8 + r;
                float v = c[t][r];
                C[(long)rowg * HID + colg] = v;                  // raw xw
                D[(long)rowg * HID + colg] = bb + v * sn2[r];    // bias + self-loop
            }
        }
    }
}

// ---------------------------------------------------------------------------
// LayerNorm over rows of A [NN,128]; writes normalized f32 in place and bf16.
// ---------------------------------------------------------------------------
__global__ void k_layernorm(float* __restrict__ A, bf16* __restrict__ hb,
                            const float* __restrict__ g, const float* __restrict__ b) {
    int n = blockIdx.x, t = threadIdx.x;
    __shared__ float s1[HID], s2[HID];
    float v = A[(long)n * HID + t];
    s1[t] = v; s2[t] = v * v;
    __syncthreads();
    for (int o = HID / 2; o > 0; o >>= 1) {
        if (t < o) { s1[t] += s1[t + o]; s2[t] += s2[t + o]; }
        __syncthreads();
    }
    float mean = s1[0] * (1.0f / HID);
    float var  = s2[0] * (1.0f / HID) - mean * mean;
    float y = (v - mean) * rsqrtf(var + 1e-5f) * g[t] + b[t];
    A[(long)n * HID + t] = y;
    hb[(long)n * HID + t] = (bf16)y;
}

// ---------------------------------------------------------------------------
// GCN degree / normalization
// ---------------------------------------------------------------------------
__global__ void k_deg_init(float* deg) {
    int i = blockIdx.x * 256 + threadIdx.x;
    if (i < NN) deg[i] = 1.0f;                 // self-loop
}
__global__ void k_deg_count(const int* __restrict__ ei, float* deg) {
    int e = blockIdx.x * 256 + threadIdx.x;
    if (e < EE) atomAddF(deg + ei[EE + e], 1.0f);   // col = target
}
__global__ void k_dis(const float* __restrict__ deg, float* __restrict__ dis) {
    int i = blockIdx.x * 256 + threadIdx.x;
    if (i < NN) dis[i] = rsqrtf(deg[i]);
}

// one wave per edge, lane covers 4 features (float4) -> 128 features
__global__ void k_edge_agg(const int* __restrict__ ei, const float* __restrict__ xw,
                           const float* __restrict__ dis, float* __restrict__ agg) {
    int e    = blockIdx.x * 8 + (threadIdx.x >> 5);
    int lane = threadIdx.x & 31;
    if (e >= EE) return;
    int r = ei[e], c = ei[EE + e];
    float nrm = dis[r] * dis[c];
    const float4 m = *(const float4*)(xw + (long)r * HID + lane * 4);
    float* dst = agg + (long)c * HID + lane * 4;
    atomAddF(dst + 0, m.x * nrm);
    atomAddF(dst + 1, m.y * nrm);
    atomAddF(dst + 2, m.z * nrm);
    atomAddF(dst + 3, m.w * nrm);
}

__global__ void k_relu_cast(float* __restrict__ A, bf16* __restrict__ hb) {
    int i = blockIdx.x * 256 + threadIdx.x;
    if (i < NN * HID) {
        float v = fmaxf(A[i], 0.0f);
        A[i] = v;
        hb[i] = (bf16)v;
    }
}

// ---------------------------------------------------------------------------
// Global mean pool
// ---------------------------------------------------------------------------
__global__ void k_pool_zero(float* pooled, float* cnts) {
    int i = blockIdx.x * 256 + threadIdx.x;
    if (i < BAT * HID) pooled[i] = 0.0f;
    if (i < BAT) cnts[i] = 0.0f;
}
__global__ void k_pool_acc(const float* __restrict__ h, const int* __restrict__ batch,
                           float* __restrict__ pooled) {
    int n    = blockIdx.x * 8 + (threadIdx.x >> 5);
    int lane = threadIdx.x & 31;
    if (n >= NN) return;
    int b = batch[n];
    const float4 m = *(const float4*)(h + (long)n * HID + lane * 4);
    float* dst = pooled + (long)b * HID + lane * 4;
    atomAddF(dst + 0, m.x);
    atomAddF(dst + 1, m.y);
    atomAddF(dst + 2, m.z);
    atomAddF(dst + 3, m.w);
}
__global__ void k_pool_cnt(const int* __restrict__ batch, float* cnts) {
    int n = blockIdx.x * 256 + threadIdx.x;
    if (n < NN) atomAddF(cnts + batch[n], 1.0f);
}
__global__ void k_pool_div(float* pooled, const float* cnts) {
    int i = blockIdx.x * 256 + threadIdx.x;
    if (i < BAT * HID) pooled[i] /= fmaxf(cnts[i >> 7], 1.0f);
}

// ---------------------------------------------------------------------------
// Memory attention (B=128, M=64, H=128, 8 heads x 16 dim)  — tiny, plain VALU
// ---------------------------------------------------------------------------
__global__ void k_qproj(const float* __restrict__ pooled, const float* __restrict__ ipw,
                        const float* __restrict__ ipb, float* __restrict__ q) {
    int b = blockIdx.x, o = threadIdx.x;
    const float* pr = pooled + (long)b * HID;
    const float* wr = ipw + (long)o * HID;     // Wq = in_proj_w[0:H], torch (out,in)
    float acc = ipb[o];
    for (int i = 0; i < HID; ++i) acc += pr[i] * wr[i];
    q[(long)b * HID + o] = acc;
}
__global__ void k_kvproj(const float* __restrict__ mb, const float* __restrict__ ipw,
                         const float* __restrict__ ipb,
                         float* __restrict__ k, float* __restrict__ v) {
    int m = blockIdx.x, o = threadIdx.x;
    const float* mr = mb + (long)m * HID;
    const float* wk = ipw + (long)(HID + o) * HID;
    const float* wv = ipw + (long)(2 * HID + o) * HID;
    float ka = ipb[HID + o], va = ipb[2 * HID + o];
    for (int i = 0; i < HID; ++i) { float x = mr[i]; ka += x * wk[i]; va += x * wv[i]; }
    k[(long)m * HID + o] = ka;
    v[(long)m * HID + o] = va;
}
__global__ void k_attention(const float* __restrict__ q, const float* __restrict__ k,
                            const float* __restrict__ v, float* __restrict__ attn) {
    int b  = blockIdx.x >> 3;
    int hh = blockIdx.x & 7;
    int m  = threadIdx.x;                      // 64 threads = 64 memory slots
    __shared__ float sp[MEMN], red[MEMN];
    const float* qp = q + (long)b * HID + hh * HDIM;
    const float* kp = k + (long)m * HID + hh * HDIM;
    float s = 0.0f;
#pragma unroll
    for (int d = 0; d < HDIM; ++d) s += qp[d] * kp[d];
    s *= 0.25f;                                // 1/sqrt(16)
    sp[m] = s; red[m] = s;
    __syncthreads();
    for (int o = MEMN / 2; o > 0; o >>= 1) {
        if (m < o) red[m] = fmaxf(red[m], red[m + o]);
        __syncthreads();
    }
    float mx = red[0];
    __syncthreads();
    float e = __expf(s - mx);
    sp[m] = e; red[m] = e;
    __syncthreads();
    for (int o = MEMN / 2; o > 0; o >>= 1) {
        if (m < o) red[m] += red[m + o];
        __syncthreads();
    }
    float inv = 1.0f / red[0];
    if (m < HDIM) {
        float acc = 0.0f;
        for (int j = 0; j < MEMN; ++j) acc += sp[j] * v[(long)j * HID + hh * HDIM + m];
        attn[(long)b * HID + hh * HDIM + m] = acc * inv;
    }
}
__global__ void k_outproj(const float* __restrict__ attn, const float* __restrict__ ow,
                          const float* __restrict__ ob, float* __restrict__ mem) {
    int b = blockIdx.x, o = threadIdx.x;
    const float* ar = attn + (long)b * HID;
    const float* wr = ow + (long)o * HID;
    float acc = ob[o];
    for (int i = 0; i < HID; ++i) acc += ar[i] * wr[i];
    mem[(long)b * HID + o] = acc;
}

// ---------------------------------------------------------------------------
// Classifier: [pooled, mem] (B x 256) -> 128 (ReLU) -> 4
// ---------------------------------------------------------------------------
__global__ void k_classifier(const float* __restrict__ pooled, const float* __restrict__ mem,
                             const float* __restrict__ cw1, const float* __restrict__ cb1,
                             const float* __restrict__ cw2, const float* __restrict__ cb2,
                             float* __restrict__ out) {
    int b = blockIdx.x, j = threadIdx.x;       // 128 threads
    __shared__ float t[HID];
    float acc = cb1[j];
    for (int i = 0; i < HID; ++i) acc += pooled[(long)b * HID + i] * cw1[(long)i * HID + j];
    for (int i = 0; i < HID; ++i) acc += mem[(long)b * HID + i] * cw1[(long)(HID + i) * HID + j];
    t[j] = fmaxf(acc, 0.0f);
    __syncthreads();
    if (j < CLS) {
        float o = cb2[j];
        for (int i = 0; i < HID; ++i) o += t[i] * cw2[i * CLS + j];
        out[b * CLS + j] = o;
    }
}

// ---------------------------------------------------------------------------
// Host orchestration
// ---------------------------------------------------------------------------
extern "C" void kernel_launch(void* const* d_in, const int* in_sizes, int n_in,
                              void* d_out, int out_size, void* d_ws, size_t ws_size,
                              hipStream_t stream) {
    const float* x     = (const float*)d_in[0];
    const int*   ei    = (const int*)  d_in[1];
    const int*   batch = (const int*)  d_in[2];
    const float* w1    = (const float*)d_in[3];
    const float* b1    = (const float*)d_in[4];
    const float* ln_g  = (const float*)d_in[5];
    const float* ln_b  = (const float*)d_in[6];
    const float* gw[3] = {(const float*)d_in[7], (const float*)d_in[9], (const float*)d_in[11]};
    const float* gb[3] = {(const float*)d_in[8], (const float*)d_in[10], (const float*)d_in[12]};
    const float* memb  = (const float*)d_in[13];
    const float* ipw   = (const float*)d_in[14];
    const float* ipb   = (const float*)d_in[15];
    const float* outw  = (const float*)d_in[16];
    const float* outb  = (const float*)d_in[17];
    const float* cw1   = (const float*)d_in[18];
    const float* cb1   = (const float*)d_in[19];
    const float* cw2   = (const float*)d_in[20];
    const float* cb2   = (const float*)d_in[21];
    float* out = (float*)d_out;

    char* ws = (char*)d_ws;
    size_t off = 0;
    auto take = [&](size_t bytes) -> char* {
        char* p = ws + off;
        off += (bytes + 255) & ~(size_t)255;
        return p;
    };
    bf16*  x_bf   = (bf16*) take((size_t)NN * DIN * sizeof(bf16));
    bf16*  w1t    = (bf16*) take((size_t)HID * DIN * sizeof(bf16));
    bf16*  gwt0   = (bf16*) take((size_t)HID * HID * sizeof(bf16));
    bf16*  gwt1   = (bf16*) take((size_t)HID * HID * sizeof(bf16));
    bf16*  gwt2   = (bf16*) take((size_t)HID * HID * sizeof(bf16));
    bf16*  gwt[3] = {gwt0, gwt1, gwt2};
    float* bufA   = (float*)take((size_t)NN * HID * sizeof(float));  // h / agg
    float* bufB   = (float*)take((size_t)NN * HID * sizeof(float));  // xw
    bf16*  h_bf   = (bf16*) take((size_t)NN * HID * sizeof(bf16));
    float* deg    = (float*)take((size_t)NN * sizeof(float));
    float* dis    = (float*)take((size_t)NN * sizeof(float));
    float* pooled = (float*)take((size_t)BAT * HID * sizeof(float));
    float* cnts   = (float*)take((size_t)BAT * sizeof(float));
    float* q      = (float*)take((size_t)BAT * HID * sizeof(float));
    float* kbuf   = (float*)take((size_t)MEMN * HID * sizeof(float));
    float* vbuf   = (float*)take((size_t)MEMN * HID * sizeof(float));
    float* attn   = (float*)take((size_t)BAT * HID * sizeof(float));
    float* memo   = (float*)take((size_t)BAT * HID * sizeof(float));
    (void)ws_size; (void)in_sizes; (void)n_in; (void)out_size;

    auto blocks = [](int n) { return (n + 255) / 256; };
    const int mstrips = NN / 16;                         // 3125 waves
    const int gemmGrid = (mstrips + 3) / 4;              // 4 waves / block

    // 1) bf16 casts
    k_cast_bf16<<<blocks(NN * DIN), 256, 0, stream>>>(x, x_bf, NN * DIN);
    k_transpose_cast<<<blocks(DIN * HID), 256, 0, stream>>>(w1, w1t, DIN, HID);
    for (int l = 0; l < 3; ++l)
        k_transpose_cast<<<blocks(HID * HID), 256, 0, stream>>>(gw[l], gwt[l], HID, HID);

    // 2) GCN normalization first (dis needed by fused GCN-GEMM epilogue)
    k_deg_init <<<blocks(NN), 256, 0, stream>>>(deg);
    k_deg_count<<<blocks(EE), 256, 0, stream>>>(ei, deg);
    k_dis      <<<blocks(NN), 256, 0, stream>>>(deg, dis);

    // 3) feature transform GEMM (bias+ReLU fused) then LayerNorm
    k_wmma_gemm_strip<<<gemmGrid, 128, 0, stream>>>(x_bf, w1t, bufA, nullptr,
                                                    NN, DIN, b1, nullptr, 0);
    k_layernorm<<<NN, HID, 0, stream>>>(bufA, h_bf, ln_g, ln_b);

    // 4) three GCN layers (GEMM epilogue writes xw AND bias+self-loop agg init)
    for (int l = 0; l < 3; ++l) {
        k_wmma_gemm_strip<<<gemmGrid, 128, 0, stream>>>(h_bf, gwt[l], bufB, bufA,
                                                        NN, HID, gb[l], dis, 1);
        k_edge_agg <<<EE / 8, 256, 0, stream>>>(ei, bufB, dis, bufA);
        k_relu_cast<<<blocks(NN * HID), 256, 0, stream>>>(bufA, h_bf);
    }

    // 5) global mean pool
    k_pool_zero<<<blocks(BAT * HID), 256, 0, stream>>>(pooled, cnts);
    k_pool_acc <<<NN / 8, 256, 0, stream>>>(bufA, batch, pooled);
    k_pool_cnt <<<blocks(NN), 256, 0, stream>>>(batch, cnts);
    k_pool_div <<<blocks(BAT * HID), 256, 0, stream>>>(pooled, cnts);

    // 6) memory attention
    k_qproj    <<<BAT, HID, 0, stream>>>(pooled, ipw, ipb, q);
    k_kvproj   <<<MEMN, HID, 0, stream>>>(memb, ipw, ipb, kbuf, vbuf);
    k_attention<<<BAT * NHEAD, MEMN, 0, stream>>>(q, kbuf, vbuf, attn);
    k_outproj  <<<BAT, HID, 0, stream>>>(attn, outw, outb, memo);

    // 7) classifier
    k_classifier<<<BAT, HID, 0, stream>>>(pooled, memo, cw1, cb1, cw2, cb2, out);
}
